// BitMGQA_90821378441601
// MI455X (gfx1250) — 8-bit matrix operand order repaired and checked by reading, compiled, not run on hardware
//
#include <hip/hip_runtime.h>
#include <cstdint>
#include <cstddef>


typedef __attribute__((ext_vector_type(8))) int   v8i;
typedef __attribute__((ext_vector_type(8))) float v8f;
typedef __attribute__((ext_vector_type(2))) float v2f;

#define N_MODEL  1024
#define KV_MODEL 256
#define HEAD_DIM 64
#define KV_HEADS 4
#define GROUPS   4
#define BATCH    4
#define SEQ      2048
#define TOKENS   (BATCH * SEQ)


__global__ void wabs_sum_kernel(const float* __restrict__ w,
                                float* __restrict__ accum, int n) {
  __shared__ float sdata[256];
  float s = 0.f;
  for (int i = blockIdx.x * blockDim.x + threadIdx.x; i < n;
       i += gridDim.x * blockDim.x)
    s += fabsf(w[i]);
  sdata[threadIdx.x] = s;
  __syncthreads();
  for (int o = 128; o > 0; o >>= 1) {
    if (threadIdx.x < o) sdata[threadIdx.x] += sdata[threadIdx.x + o];
    __syncthreads();
  }
  if (threadIdx.x == 0) atomicAdd(accum, sdata[0]);
}

__global__ void wquant_kernel(const float* __restrict__ w,
                              int8_t* __restrict__ wq,
                              const float* __restrict__ accum, float inv_n,
                              int n) {
  const float gamma = accum[0] * inv_n;
  const float invg = 1.f / (gamma + 1e-5f);
  for (int i = blockIdx.x * blockDim.x + threadIdx.x; i < n;
       i += gridDim.x * blockDim.x) {
    float q = rintf(w[i] * invg);
    q = fminf(fmaxf(q, -1.f), 1.f);
    wq[i] = (int8_t)q;
  }
}

__global__ void wqsum_kernel(const int8_t* __restrict__ Wq_i8,
                             int8_t* __restrict__ Wqs) {
  const int idx = blockIdx.x * blockDim.x + threadIdx.x;
  const int row = idx >> 10;
  const int k = idx & 1023;
  const int kv = row >> 6, d = row & 63;
  int s = 0;
#pragma unroll
  for (int g = 0; g < GROUPS; ++g)
    s += (int)Wq_i8[(size_t)((kv * GROUPS + g) * HEAD_DIM + d) * N_MODEL + k];
  Wqs[idx] = (int8_t)s;
}


__global__ void aquant_kernel(const float* __restrict__ x,
                              int8_t* __restrict__ q,
                              float* __restrict__ inv_scale, int IN,
                              float scale) {
  __shared__ float sdata[256];
  const int t = blockIdx.x;
  const float* row = x + (size_t)t * IN;
  float ss = 0.f, am = 0.f;
  for (int i = threadIdx.x; i < IN; i += 256) {
    float v = row[i];
    ss += v * v;
    am = fmaxf(am, fabsf(v));
  }
  sdata[threadIdx.x] = ss;
  __syncthreads();
  for (int o = 128; o > 0; o >>= 1) {
    if (threadIdx.x < o) sdata[threadIdx.x] += sdata[threadIdx.x + o];
    __syncthreads();
  }
  const float nrm = sqrtf(sdata[0]);
  __syncthreads();
  sdata[threadIdx.x] = am;
  __syncthreads();
  for (int o = 128; o > 0; o >>= 1) {
    if (threadIdx.x < o)
      sdata[threadIdx.x] = fmaxf(sdata[threadIdx.x], sdata[threadIdx.x + o]);
    __syncthreads();
  }
  const float amax_raw = sdata[0];
  const float f = scale / fmaxf(nrm, 1e-12f);
  const float amax = fmaxf(amax_raw * f, 1e-5f);
  const float a_scale = 127.f / amax;
  if (threadIdx.x == 0) inv_scale[t] = amax / 127.f;
  const float ff = f * a_scale;
  for (int i = threadIdx.x; i < IN; i += 256) {
    float v = rintf(row[i] * ff);
    v = fminf(fmaxf(v, -127.f), 127.f);
    q[(size_t)t * IN + i] = (int8_t)v;
  }
}


__global__ void __launch_bounds__(256) gemm_iu8_kernel(
    const int8_t* __restrict__ A, const int8_t* __restrict__ W,
    const float* __restrict__ inv_a, const float* __restrict__ gaccum,
    float inv_n, float* __restrict__ C, int N, int K, int nt) {
  const int lane = threadIdx.x & 31;
  const int wave = threadIdx.x >> 5;
  const int half = lane >> 4;
  const int l    = lane & 15;
  const int m0 = blockIdx.y * 32 + (wave >> 2) * 16;
  const int n0 = blockIdx.x * 64 + (wave & 3) * 16;
  const int8_t* arow = A + (size_t)(m0 + l) * K;
  const int8_t* brow = W + (size_t)(n0 + l) * K;
  v8i acc = {0, 0, 0, 0, 0, 0, 0, 0};
  for (int k = 0; k < K; k += 64) {
    __builtin_prefetch(arow + k + 64, 0, 3);
    __builtin_prefetch(brow + k + 64, 0, 3);
    v8i a, b;
#pragma unroll
    for (int j = 0; j < 8; ++j) {
      const int ka = k + (((j >> 1) & 1) * 16) + ((j & 1) * 4) + half * 8 + ((j >> 2) * 32);
      a[j] = *(const int*)(arow + ka);
      const int kb = k + (((j >> 1) & 1) * 16) + ((j & 1) * 4) + half * 8 + ((j >> 2) * 32);
      b[j] = *(const int*)(brow + kb);
    }
    acc = __builtin_amdgcn_wmma_i32_16x16x64_iu8(true, a, true, b, acc,
                                                 false, false);
  }
  const float gamma = gaccum[0] * inv_n;
  const int mb = m0 + half * 8;
  if (nt) {
#pragma unroll
    for (int r = 0; r < 8; ++r) {
      const int m = mb + r;
      __builtin_nontemporal_store((float)acc[r] * gamma * inv_a[m],
                                  &C[(size_t)m * N + n0 + l]);
    }
  } else {
#pragma unroll
    for (int r = 0; r < 8; ++r) {
      const int m = mb + r;
      C[(size_t)m * N + n0 + l] = (float)acc[r] * gamma * inv_a[m];
    }
  }
}


__global__ void __launch_bounds__(256) fused_attn_kernel(
    const float* __restrict__ Qs, const float* __restrict__ Kf,
    const float* __restrict__ Vf, float* __restrict__ P,
    float* __restrict__ X) {
  extern __shared__ float lds[];
  const int lane = threadIdx.x & 31;
  const int wave = threadIdx.x >> 5;
  const int half = lane >> 4;
  const int l    = lane & 15;
  const int bz = blockIdx.y;
  const int b = bz >> 2, kv = bz & 3;
  const int n0 = blockIdx.x * 32;

  const int wm = (wave >> 2) * 16;
  {
    const float* qrow =
        Qs + (size_t)(b * SEQ + n0 + wm + l) * KV_MODEL + kv * HEAD_DIM;
    v2f afrag[16];
#pragma unroll
    for (int d4 = 0; d4 < 16; ++d4)
      afrag[d4] = *(const v2f*)(qrow + d4 * 4 + half * 2);
    for (int i = 0; i < 32; ++i) {
      const int ts = ((wave & 3) + 4 * i) * 16;
      const float* krow =
          Kf + (size_t)(b * SEQ + ts + l) * KV_MODEL + kv * HEAD_DIM;
      v8f acc = {0.f, 0.f, 0.f, 0.f, 0.f, 0.f, 0.f, 0.f};
#pragma unroll
      for (int d4 = 0; d4 < 16; ++d4) {
        v2f bb = *(const v2f*)(krow + d4 * 4 + half * 2);
        acc = __builtin_amdgcn_wmma_f32_16x16x4_f32(false, afrag[d4], false, bb,
                                                    (short)0, acc, false, false);
      }
      const int mb = wm + half * 8;
#pragma unroll
      for (int r = 0; r < 8; ++r)
        lds[(size_t)(mb + r) * SEQ + ts + l] = acc[r] * 0.125f;
    }
  }
  __syncthreads();

  for (int r = 0; r < 4; ++r) {
    const int row = wave * 4 + r;
    float* prow = lds + (size_t)row * SEQ;
    float m = -3.4e38f;
    for (int i = lane; i < SEQ; i += 32) m = fmaxf(m, prow[i]);
#pragma unroll
    for (int o = 16; o > 0; o >>= 1) m = fmaxf(m, __shfl_xor(m, o, 32));
    float sum = 0.f;
    for (int i = lane; i < SEQ; i += 32) {
      const float e = __expf(prow[i] - m);
      prow[i] = e;
      sum += e;
    }
#pragma unroll
    for (int o = 16; o > 0; o >>= 1) sum += __shfl_xor(sum, o, 32);
    const float inv = 1.f / sum;
    float* gp = P + ((size_t)bz * SEQ + n0 + row) * SEQ;
    for (int i = lane; i < SEQ; i += 32) {
      const float p = prow[i] * inv;
      prow[i] = p;
      __builtin_nontemporal_store(p, gp + i);
    }
  }
  __syncthreads();

  {
    const int d0 = (wave & 3) * 16;
    const float* arow = lds + (size_t)(wm + l) * SEQ;
    const float* vbase =
        Vf + (size_t)(b * SEQ) * KV_MODEL + kv * HEAD_DIM + d0 + l;
    v8f acc = {0.f, 0.f, 0.f, 0.f, 0.f, 0.f, 0.f, 0.f};
    for (int s = 0; s < SEQ; s += 4) {
      v2f a = *(const v2f*)(arow + s + half * 2);
      v2f bb;
      bb.x = vbase[(size_t)(s + half * 2 + 0) * KV_MODEL];
      bb.y = vbase[(size_t)(s + half * 2 + 1) * KV_MODEL];
      acc = __builtin_amdgcn_wmma_f32_16x16x4_f32(false, a, false, bb, (short)0,
                                                  acc, false, false);
    }
    const int mb = wm + half * 8;
#pragma unroll
    for (int r = 0; r < 8; ++r)
      X[((size_t)(b * SEQ) + n0 + mb + r) * KV_MODEL + kv * HEAD_DIM + d0 + l] =
          acc[r];
  }
}


extern "C" void kernel_launch(void* const* d_in, const int* in_sizes, int n_in,
                              void* d_out, int out_size, void* d_ws,
                              size_t ws_size, hipStream_t stream) {
  (void)in_sizes; (void)n_in; (void)out_size; (void)ws_size;
  const float* q_in = (const float*)d_in[0];
  const float* k_in = (const float*)d_in[1];
  const float* v_in = (const float*)d_in[2];
  const float* Wq = (const float*)d_in[3];
  const float* Wk = (const float*)d_in[4];
  const float* Wv = (const float*)d_in[5];
  const float* Wo = (const float*)d_in[6];

  float* outp  = (float*)d_out;
  float* attnp = outp + (size_t)TOKENS * N_MODEL;

  char* ws = (char*)d_ws;
  float*  gacc   = (float*)ws;
  float*  inva   = (float*)(ws + 256);
  int8_t* wq_i8  = (int8_t*)(ws + (1u << 16));
  int8_t* wk_i8  = wq_i8 + (size_t)N_MODEL * N_MODEL;
  int8_t* wv_i8  = wk_i8 + (size_t)KV_MODEL * N_MODEL;
  int8_t* wo_i8  = wv_i8 + (size_t)KV_MODEL * N_MODEL;
  int8_t* wqs_i8 = wo_i8 + (size_t)N_MODEL * KV_MODEL;
  int8_t* act_i8 = (int8_t*)(ws + (4ull  << 20));
  float*  Qs     = (float*)(ws + (12ull << 20));
  float*  Kf     = (float*)(ws + (20ull << 20));
  float*  Vf     = (float*)(ws + (28ull << 20));
  float*  Xf     = (float*)(ws + (36ull << 20));

  hipMemsetAsync(gacc, 0, 4 * sizeof(float), stream);

  wabs_sum_kernel<<<512, 256, 0, stream>>>(Wq, gacc + 0, N_MODEL * N_MODEL);
  wabs_sum_kernel<<<256, 256, 0, stream>>>(Wk, gacc + 1, KV_MODEL * N_MODEL);
  wabs_sum_kernel<<<256, 256, 0, stream>>>(Wv, gacc + 2, KV_MODEL * N_MODEL);
  wabs_sum_kernel<<<256, 256, 0, stream>>>(Wo, gacc + 3, N_MODEL * KV_MODEL);
  wquant_kernel<<<512, 256, 0, stream>>>(Wq, wq_i8, gacc + 0,
                                         1.f / (N_MODEL * N_MODEL), N_MODEL * N_MODEL);
  wquant_kernel<<<256, 256, 0, stream>>>(Wk, wk_i8, gacc + 1,
                                         1.f / (KV_MODEL * N_MODEL), KV_MODEL * N_MODEL);
  wquant_kernel<<<256, 256, 0, stream>>>(Wv, wv_i8, gacc + 2,
                                         1.f / (KV_MODEL * N_MODEL), KV_MODEL * N_MODEL);
  wquant_kernel<<<256, 256, 0, stream>>>(Wo, wo_i8, gacc + 3,
                                         1.f / (N_MODEL * KV_MODEL), N_MODEL * KV_MODEL);
  wqsum_kernel<<<(KV_MODEL * N_MODEL) / 256, 256, 0, stream>>>(wq_i8, wqs_i8);

  const float sc1024 = 0.03125f;
  const float sc256  = 0.0625f;

  aquant_kernel<<<TOKENS, 256, 0, stream>>>(q_in, act_i8, inva, N_MODEL, sc1024);
  gemm_iu8_kernel<<<dim3(KV_MODEL / 64, TOKENS / 32), 256, 0, stream>>>(
      act_i8, wqs_i8, inva, gacc + 0, 1.f / (N_MODEL * N_MODEL), Qs,
      KV_MODEL, N_MODEL, 0);

  aquant_kernel<<<TOKENS, 256, 0, stream>>>(k_in, act_i8, inva, N_MODEL, sc1024);
  gemm_iu8_kernel<<<dim3(KV_MODEL / 64, TOKENS / 32), 256, 0, stream>>>(
      act_i8, wk_i8, inva, gacc + 1, 1.f / (KV_MODEL * N_MODEL), Kf,
      KV_MODEL, N_MODEL, 0);

  aquant_kernel<<<TOKENS, 256, 0, stream>>>(v_in, act_i8, inva, N_MODEL, sc1024);
  gemm_iu8_kernel<<<dim3(KV_MODEL / 64, TOKENS / 32), 256, 0, stream>>>(
      act_i8, wv_i8, inva, gacc + 2, 1.f / (KV_MODEL * N_MODEL), Vf,
      KV_MODEL, N_MODEL, 0);

  fused_attn_kernel<<<dim3(SEQ / 32, BATCH * KV_HEADS), 256,
                      32 * SEQ * sizeof(float), stream>>>(Qs, Kf, Vf, attnp, Xf);

  aquant_kernel<<<TOKENS, 256, 0, stream>>>(Xf, act_i8, inva, KV_MODEL, sc256);
  gemm_iu8_kernel<<<dim3(N_MODEL / 64, TOKENS / 32), 256, 0, stream>>>(
      act_i8, wo_i8, inva, gacc + 3, 1.f / (N_MODEL * KV_MODEL), outp,
      N_MODEL, KV_MODEL, 1);
}
